// MP_GAT_58755152609379
// MI455X (gfx1250) — matrix operand order repaired and checked by reading, not compiled, not run on hardware
//
#include <hip/hip_runtime.h>
#include <hip/hip_bf16.h>

typedef __attribute__((ext_vector_type(16))) __bf16 v16bf;
typedef __attribute__((ext_vector_type(8)))  float  v8f;
typedef __attribute__((ext_vector_type(4)))  float  f4;

#define B_ 8
#define S_ 64
#define N_ 512
#define E_ 32

// ---------------------------------------------------------------------------
// Kernel 1: tij[b][n][0:64] = [ti | tj],  ti = v @ Wi^T, tj = v @ Wj^T
// One wave per 16(n) x 16(e') tile, K=64 via two K=32 bf16 WMMAs.
// 1024 waves exactly -> <<<256,128>>> (no tail, EXEC all ones).
// ---------------------------------------------------------------------------
__global__ void __launch_bounds__(128) mpgat_gemm1(const float* __restrict__ x,
                                                   const float* __restrict__ W,
                                                   float* __restrict__ tij) {
    const int tid  = blockIdx.x * 128 + threadIdx.x;
    const int wid  = tid >> 5;
    const int lane = threadIdx.x & 31;
    const int l    = lane & 15;
    const int hi   = lane >> 4;

    const int etile = wid & 3;          // 4 tiles of 16 over e' = 0..63
    const int ntile = (wid >> 2) & 31;  // 32 tiles of 16 over n
    const int b     = wid >> 7;         // batch

    const int ep    = etile * 16 + l;   // 0..63 (col of combined [Wi|Wj])
    const int wrow  = ep & 31;          // row of W
    const int wcol0 = (ep >= 32) ? S_ : 0;
    const int ncol  = ntile * 16 + l;   // A-matrix row (n index)

    v8f c = {};
#pragma unroll
    for (int kk = 0; kk < S_; kk += 32) {
        v16bf a, bm;
        // A fragment: A[n,s] = x[b,s,n]  (strided gather over s)
#pragma unroll
        for (int i = 0; i < 16; ++i) {
            const int ki = 16 * hi + i;
            const int s  = kk + ki;
            a[i] = (__bf16)x[b * (S_ * N_) + s * N_ + ncol];
        }
        // B fragment: 16 contiguous columns of one W row
        const float* wp = W + wrow * (2 * S_) + wcol0 + kk + 16 * hi;
#pragma unroll
        for (int q = 0; q < 4; ++q) {
            f4 v = ((const f4*)wp)[q];
#pragma unroll
            for (int u = 0; u < 4; ++u) bm[4 * q + u] = (__bf16)v[u];
        }
        c = __builtin_amdgcn_wmma_f32_16x16x32_bf16(false, a, false, bm,
                                                    (short)0, c, false, false);
    }
    float* dst = tij + b * (N_ * 64) + (ntile * 16 + 8 * hi) * 64 + etile * 16 + l;
#pragma unroll
    for (int r = 0; r < 8; ++r) dst[r * 64] = c[r];
}

// ---------------------------------------------------------------------------
// Kernel 2: e[b,i,j] = sum_e lrelu(ti[b,i,e] + tj[b,j,e] + b_w[e]) * Wa[e] + b_a
// One block per (b,i); ti row + b_w + Wa cached in LDS; tj streamed via float4.
// ---------------------------------------------------------------------------
__global__ void __launch_bounds__(256) mpgat_edge(const float* __restrict__ tij,
                                                  const float* __restrict__ b_w,
                                                  const float* __restrict__ Wa,
                                                  const float* __restrict__ b_a,
                                                  float* __restrict__ att) {
    __shared__ float s_ti[E_], s_bw[E_], s_wa[E_];
    const int b = blockIdx.x >> 9;
    const int i = blockIdx.x & (N_ - 1);
    const int t = threadIdx.x;
    if (t < E_) {
        s_ti[t] = tij[b * (N_ * 64) + i * 64 + t];
        s_bw[t] = b_w[t];
        s_wa[t] = Wa[t];
    }
    __syncthreads();
    const float ba = b_a[0];
#pragma unroll
    for (int rep = 0; rep < 2; ++rep) {
        const int j = t + rep * 256;
        const f4* tjp = (const f4*)(tij + b * (N_ * 64) + j * 64 + E_);
        float acc = 0.f;
#pragma unroll
        for (int q = 0; q < 8; ++q) {
            f4 v = tjp[q];
#pragma unroll
            for (int u = 0; u < 4; ++u) {
                const int e = 4 * q + u;
                const float pre = s_ti[e] + v[u] + s_bw[e];
                acc += (pre > 0.f ? pre : 0.2f * pre) * s_wa[e];
            }
        }
        att[(size_t)b * (N_ * N_) + (size_t)i * N_ + j] = acc + ba;
    }
}

// ---------------------------------------------------------------------------
// Kernel 3: softmax over j (row length 512), one wave32 per row,
// 16 elements/lane, __shfl_xor reductions (wave32-native).
// ---------------------------------------------------------------------------
__global__ void __launch_bounds__(128) mpgat_softmax(float* __restrict__ att) {
    const int row  = blockIdx.x * 4 + (threadIdx.x >> 5);
    const int lane = threadIdx.x & 31;
    float* p = att + (size_t)row * N_ + lane * 16;
    f4 v[4];
#pragma unroll
    for (int q = 0; q < 4; ++q) v[q] = ((f4*)p)[q];
    float m = -3.0e38f;
#pragma unroll
    for (int q = 0; q < 4; ++q)
#pragma unroll
        for (int u = 0; u < 4; ++u) m = fmaxf(m, v[q][u]);
#pragma unroll
    for (int off = 16; off >= 1; off >>= 1) m = fmaxf(m, __shfl_xor(m, off, 32));
    float sum = 0.f;
#pragma unroll
    for (int q = 0; q < 4; ++q)
#pragma unroll
        for (int u = 0; u < 4; ++u) {
            const float ev = __expf(v[q][u] - m);
            v[q][u] = ev;
            sum += ev;
        }
#pragma unroll
    for (int off = 16; off >= 1; off >>= 1) sum += __shfl_xor(sum, off, 32);
    const float inv = 1.f / sum;
#pragma unroll
    for (int q = 0; q < 4; ++q) {
#pragma unroll
        for (int u = 0; u < 4; ++u) v[q][u] *= inv;
        ((f4*)p)[q] = v[q];
    }
}

// ---------------------------------------------------------------------------
// Kernel 4: h[b,s,i] = sigmoid( sum_j att[b,i,j] * x[b,s,j] )
// One wave per 16(i) x 16(s) tile, K=512 via 16 bf16 WMMAs.
// A-frag: two contiguous 8-float runs per lane; B-frag: 16 contiguous floats
// (K index j is the fast axis of x since v[b,j,s] = x[b,s,j]).
// 1024 waves exactly -> <<<256,128>>>.
// ---------------------------------------------------------------------------
__global__ void __launch_bounds__(128) mpgat_gemm2(const float* __restrict__ att,
                                                   const float* __restrict__ x,
                                                   float* __restrict__ h_out) {
    const int tid  = blockIdx.x * 128 + threadIdx.x;
    const int wid  = tid >> 5;
    const int lane = threadIdx.x & 31;
    const int l    = lane & 15;
    const int hi   = lane >> 4;

    const int stile = wid & 3;
    const int itile = (wid >> 2) & 31;
    const int b     = wid >> 7;

    const int irow = itile * 16 + l;
    const int scol = stile * 16 + l;
    const float* arow = att + (size_t)b * (N_ * N_) + (size_t)irow * N_;
    const float* brow = x + b * (S_ * N_) + scol * N_;

    v8f c = {};
#pragma unroll 4
    for (int kt = 0; kt < 16; ++kt) {
        v16bf a, bm;
        const float* ap = arow + kt * 32 + 16 * hi;
        f4 a0 = ((const f4*)ap)[0];
        f4 a1 = ((const f4*)ap)[1];
        f4 a2 = ((const f4*)(ap + 8))[0];
        f4 a3 = ((const f4*)(ap + 8))[1];
#pragma unroll
        for (int u = 0; u < 4; ++u) {
            a[u]      = (__bf16)a0[u];
            a[4 + u]  = (__bf16)a1[u];
            a[8 + u]  = (__bf16)a2[u];
            a[12 + u] = (__bf16)a3[u];
        }
        const float* bp = brow + kt * 32 + 16 * hi;
#pragma unroll
        for (int q = 0; q < 4; ++q) {
            f4 v = ((const f4*)bp)[q];
#pragma unroll
            for (int u = 0; u < 4; ++u) bm[4 * q + u] = (__bf16)v[u];
        }
        c = __builtin_amdgcn_wmma_f32_16x16x32_bf16(false, a, false, bm,
                                                    (short)0, c, false, false);
    }
#pragma unroll
    for (int r = 0; r < 8; ++r) {
        const int i_out = itile * 16 + 8 * hi + r;
        const float hval = 1.f / (1.f + __expf(-c[r]));
        h_out[b * (S_ * N_) + scol * N_ + i_out] = hval;
    }
}

// ---------------------------------------------------------------------------
extern "C" void kernel_launch(void* const* d_in, const int* in_sizes, int n_in,
                              void* d_out, int out_size, void* d_ws, size_t ws_size,
                              hipStream_t stream) {
    (void)in_sizes; (void)n_in; (void)out_size; (void)ws_size;
    const float* x   = (const float*)d_in[0];
    const float* W   = (const float*)d_in[1];
    const float* b_w = (const float*)d_in[2];
    const float* Wa  = (const float*)d_in[3];
    const float* b_a = (const float*)d_in[4];

    float* h_out = (float*)d_out;                 // (B,S,N) = 262144 floats
    float* att   = (float*)d_out + B_ * S_ * N_;  // (B,N,N) = 2097152 floats
    float* tij   = (float*)d_ws;                  // (B,N,64) = 1 MB scratch

    mpgat_gemm1<<<256, 128, 0, stream>>>(x, W, tij);
    mpgat_edge<<<B_ * N_, 256, 0, stream>>>(tij, b_w, Wa, b_a, att);
    mpgat_softmax<<<(B_ * N_) / 4, 128, 0, stream>>>(att);
    mpgat_gemm2<<<256, 128, 0, stream>>>(att, x, h_out);
}
